// MC_unPool_12343736009109
// MI455X (gfx1250) — matrix operand order repaired and checked by reading, not compiled, not run on hardware
//
#include <hip/hip_runtime.h>
#include <hip/hip_bf16.h>
#include <math.h>

typedef __attribute__((ext_vector_type(16))) __bf16 v16bf;
typedef __attribute__((ext_vector_type(8)))  __bf16 v8bf;
typedef __attribute__((ext_vector_type(4)))  __bf16 v4bf;
typedef __attribute__((ext_vector_type(8)))  float  v8f;

#define EPSV 1e-7f
#define N_NODES 8192
#define F_DIM 64
#define H_DIM 16
#define K_CL 1024

// ---------------- block reductions ----------------
__device__ inline float blockReduceSum(float v, float* sh) {
    int tid = threadIdx.x;
    sh[tid] = v; __syncthreads();
    for (int s = blockDim.x >> 1; s > 0; s >>= 1) {
        if (tid < s) sh[tid] += sh[tid + s];
        __syncthreads();
    }
    float r = sh[0]; __syncthreads();
    return r;
}
__device__ inline float blockReduceMax(float v, float* sh) {
    int tid = threadIdx.x;
    sh[tid] = v; __syncthreads();
    for (int s = blockDim.x >> 1; s > 0; s >>= 1) {
        if (tid < s) sh[tid] = fmaxf(sh[tid], sh[tid + s]);
        __syncthreads();
    }
    float r = sh[0]; __syncthreads();
    return r;
}

// ---------------- A f32 -> bf16 + degree rowsum (read A once) ----------------
__global__ __launch_bounds__(256) void k_convert_rowsum(
    const float* __restrict__ A, __bf16* __restrict__ Ab, float* __restrict__ dsum)
{
    __shared__ float sh[256];
    const int row = blockIdx.x;
    const int tid = threadIdx.x;
    const float4* Ar = (const float4*)(A + (size_t)row * N_NODES);
    float s = 0.f;
    #pragma unroll
    for (int i = 0; i < (N_NODES / 4) / 256; ++i) {
        int idx = tid + i * 256;
        float4 v = Ar[idx];
        s += v.x + v.y + v.z + v.w;
        v4bf b;
        b[0] = (__bf16)v.x; b[1] = (__bf16)v.y; b[2] = (__bf16)v.z; b[3] = (__bf16)v.w;
        *(v4bf*)(Ab + (size_t)row * N_NODES + (size_t)idx * 4) = b;
    }
    s = blockReduceSum(s, sh);
    if (tid == 0) dsum[row] = s;
}

// ---------------- assignment MLP + softmax: S (bf16, coalesced), X bf16, row sum(S*S) ----------------
__global__ __launch_bounds__(256) void k_assign(
    const float* __restrict__ X, const float* __restrict__ W1, const float* __restrict__ b1,
    const float* __restrict__ W2, const float* __restrict__ b2,
    __bf16* __restrict__ Sb, __bf16* __restrict__ Xb, float* __restrict__ rowSS)
{
    __shared__ float xr[F_DIM];
    __shared__ float h[H_DIM];
    __shared__ float sh[256];
    const int row = blockIdx.x;
    const int tid = threadIdx.x;

    if (tid < F_DIM) {
        float v = X[(size_t)row * F_DIM + tid];
        xr[tid] = v;
        Xb[(size_t)row * F_DIM + tid] = (__bf16)v;
    }
    __syncthreads();
    if (tid < H_DIM) {
        float a = b1[tid];
        #pragma unroll
        for (int f = 0; f < F_DIM; ++f) a += xr[f] * W1[f * H_DIM + tid];
        h[tid] = fmaxf(a, 0.f);
    }
    __syncthreads();

    float lg[4];
    float mx = -1e30f;
    #pragma unroll
    for (int j = 0; j < 4; ++j) {
        int col = tid + j * 256;
        float a = b2[col];
        #pragma unroll
        for (int l = 0; l < H_DIM; ++l) a += h[l] * W2[l * K_CL + col];
        lg[j] = a;
        mx = fmaxf(mx, a);
    }
    mx = blockReduceMax(mx, sh);
    float es = 0.f;
    #pragma unroll
    for (int j = 0; j < 4; ++j) { lg[j] = __expf(lg[j] - mx); es += lg[j]; }
    es = blockReduceSum(es, sh);
    float inv = 1.f / es;
    float ss = 0.f;
    #pragma unroll
    for (int j = 0; j < 4; ++j) {
        int col = tid + j * 256;
        float s = lg[j] * inv;
        ss += s * s;
        Sb[(size_t)row * K_CL + col] = (__bf16)s;
    }
    ss = blockReduceSum(ss, sh);
    if (tid == 0) rowSS[row] = ss;
}

// ---------------- bf16 tiled transpose: src[R][C] -> dst[C][R] (R,C multiples of 64) ----------------
__global__ __launch_bounds__(256) void k_transpose_bf16(
    const __bf16* __restrict__ src, __bf16* __restrict__ dst, int R, int C)
{
    __shared__ __align__(16) __bf16 t[64][72];   // 144B pitch: 36-bank shift, 16B-aligned rows
    const int r0 = blockIdx.x * 64;
    const int c0 = blockIdx.y * 64;
    const int tid = threadIdx.x;
    #pragma unroll
    for (int i = 0; i < 2; ++i) {
        int idx = tid + i * 256;           // 0..511
        int r = idx >> 3;                  // 0..63
        int c = (idx & 7) << 3;            // 0,8,..,56
        v8bf v = *(const v8bf*)(src + (size_t)(r0 + r) * C + c0 + c);
        #pragma unroll
        for (int j = 0; j < 8; ++j) t[c + j][r] = v[j];
    }
    __syncthreads();
    #pragma unroll
    for (int i = 0; i < 2; ++i) {
        int idx = tid + i * 256;
        int c = idx >> 3;
        int r = (idx & 7) << 3;
        v8bf v = *(const v8bf*)&t[c][r];
        *(v8bf*)(dst + (size_t)(c0 + c) * R + r0 + r) = v;
    }
}

// ---------------- bf16 WMMA GEMM: C[M,N] = A[M,K] @ Bt[N,K]^T, f32 accum ----------------
// Both operands K-contiguous; double-buffered LDS; 8 waves, 128x128 tile, K-tile 32.
// NG:     guard B rows against N (only needed when N < 128)
// OUTMODE 0: f32 C[M,N] (col-guarded)   1: bf16 transposed Ct[N,M], packed v8bf stores
template<bool NG, int OUTMODE>
__global__ __launch_bounds__(256) void k_gemm_bf16(
    const __bf16* __restrict__ A, const __bf16* __restrict__ Bt,
    float* __restrict__ Cf, __bf16* __restrict__ Ct,
    int M, int N, int K)
{
    __shared__ __align__(16) __bf16 Al[2][128][32];
    __shared__ __align__(16) __bf16 Bl[2][128][32];

    const int tid  = threadIdx.x;
    const int lane = tid & 31;
    const int wave = tid >> 5;
    const int wm = wave >> 2;              // 0..1
    const int wn = wave & 3;               // 0..3
    const int m0 = blockIdx.x * 128;
    const int n0 = blockIdx.y * 128;
    const int lm = lane & 15;
    const int kA = (lane < 16) ? 0 : 8;    // A frag: lanes 16..31 hold K=8..15, 24..31
    const int kB = (lane < 16) ? 0 : 8;    // B frag: lanes 16..31 hold K=8..15, 24..31

    // per-thread tile-load coordinates (same scheme for A and B: 2 x v8bf each)
    const int lr0 = tid >> 2;                    // 0..63
    const int lr1 = (tid + 256) >> 2;            // 64..127
    const int lc  = (tid & 3) << 3;              // 0,8,16,24

    auto loadTile = [&](int buf, int k0) {
        *(v8bf*)&Al[buf][lr0][lc] = *(const v8bf*)(A + (size_t)(m0 + lr0) * K + k0 + lc);
        *(v8bf*)&Al[buf][lr1][lc] = *(const v8bf*)(A + (size_t)(m0 + lr1) * K + k0 + lc);
        v8bf z = {};
        v8bf b0 = (!NG || (n0 + lr0) < N) ? *(const v8bf*)(Bt + (size_t)(n0 + lr0) * K + k0 + lc) : z;
        v8bf b1 = (!NG || (n0 + lr1) < N) ? *(const v8bf*)(Bt + (size_t)(n0 + lr1) * K + k0 + lc) : z;
        *(v8bf*)&Bl[buf][lr0][lc] = b0;
        *(v8bf*)&Bl[buf][lr1][lc] = b1;
    };

    v8f acc[4][2] = {};

    loadTile(0, 0);
    int buf = 0;
    for (int k0 = 0; k0 < K; k0 += 32) {
        __syncthreads();                       // tile[buf] ready; prior reads of tile[buf^1] done
        if (k0 + 32 < K) loadTile(buf ^ 1, k0 + 32);

        // Preload ALL fragments into distinct registers first, so the 8 WMMAs
        // issue back-to-back without per-step s_wait_dscnt 0 stalls (WMMA on
        // 16-bit data co-executes with VALU/LDS).
        union Frag { v8bf h[2]; v16bf v; };
        Frag bfrag[2], afrag[4];
        #pragma unroll
        for (int ni = 0; ni < 2; ++ni) {
            int rB = wn * 32 + ni * 16 + lm;
            bfrag[ni].h[0] = *(const v8bf*)&Bl[buf][rB][kB];
            bfrag[ni].h[1] = *(const v8bf*)&Bl[buf][rB][kB + 16];
        }
        #pragma unroll
        for (int mi = 0; mi < 4; ++mi) {
            int rA = wm * 64 + mi * 16 + lm;
            afrag[mi].h[0] = *(const v8bf*)&Al[buf][rA][kA];        // K = kA..kA+7
            afrag[mi].h[1] = *(const v8bf*)&Al[buf][rA][kA + 16];   // K = kA+16..kA+23
        }
        #pragma unroll
        for (int mi = 0; mi < 4; ++mi) {
            #pragma unroll
            for (int ni = 0; ni < 2; ++ni) {
                acc[mi][ni] = __builtin_amdgcn_wmma_f32_16x16x32_bf16(
                    false, afrag[mi].v, false, bfrag[ni].v, (short)0, acc[mi][ni], false, false);
            }
        }
        buf ^= 1;
    }

    // C/D layout: lanes 0..15 -> M=v, lanes 16..31 -> M=v+8; N = lane%16
    const int rsel = (lane < 16) ? 0 : 8;
    #pragma unroll
    for (int mi = 0; mi < 4; ++mi) {
        #pragma unroll
        for (int ni = 0; ni < 2; ++ni) {
            int cn = n0 + wn * 32 + ni * 16 + lm;
            int rb = m0 + wm * 64 + mi * 16 + rsel;
            if (OUTMODE == 0) {
                if (cn < N) {
                    #pragma unroll
                    for (int v = 0; v < 8; ++v)
                        Cf[(size_t)(rb + v) * N + cn] = acc[mi][ni][v];
                }
            } else {
                v8bf p;
                #pragma unroll
                for (int v = 0; v < 8; ++v) p[v] = (__bf16)acc[mi][ni][v];
                *(v8bf*)(Ct + (size_t)cn * M + rb) = p;   // 8 consecutive M -> one b128
            }
        }
    }
}

// ---------------- scalar losses + I_pooled zeros ----------------
// ortho = || SS/||SS|| - I/sqrt(k) ||_F = sqrt(2 - 2*tr(SS)/(||SS||_F*sqrt(k)))
__global__ __launch_bounds__(1024) void k_scalars(
    const float* __restrict__ Arec, const float* __restrict__ SSm,
    const float* __restrict__ dsum, const float* __restrict__ rowSS,
    float* __restrict__ out_cut, float* __restrict__ out_ortho, float* __restrict__ out_ip)
{
    __shared__ float sh[1024];
    const int tid = threadIdx.x;
    float tA = 0.f, tS = 0.f, q = 0.f, dn = 0.f;
    for (int i = tid; i < K_CL; i += 1024) {
        tA += Arec[(size_t)i * K_CL + i];
        tS += SSm [(size_t)i * K_CL + i];
    }
    for (size_t idx = tid; idx < (size_t)K_CL * K_CL; idx += 1024) {
        float v = SSm[idx];
        q += v * v;
    }
    for (int n = tid; n < N_NODES; n += 1024) dn += dsum[n] * rowSS[n];
    tA = blockReduceSum(tA, sh);
    tS = blockReduceSum(tS, sh);
    q  = blockReduceSum(q,  sh);
    dn = blockReduceSum(dn, sh);
    if (tid == 0) {
        *out_cut = -(tA / (dn + EPSV));
        float nrm = sqrtf(q);
        float o2 = 2.f - 2.f * tS / (nrm * sqrtf((float)K_CL));
        *out_ortho = sqrtf(fmaxf(o2, 0.f));
    }
    out_ip[tid] = 0.f;   // segment-mean of all-zero I, repeated k times
}

// ---------------- A_rec degree (diag excluded) -> dis = deg^-0.5 ----------------
__global__ __launch_bounds__(256) void k_dis(
    const float* __restrict__ Arec, float* __restrict__ dis)
{
    __shared__ float sh[256];
    const int i = blockIdx.x;
    const int tid = threadIdx.x;
    float s = 0.f;
    for (int j = tid; j < K_CL; j += 256)
        if (j != i) s += Arec[(size_t)i * K_CL + j];
    s = blockReduceSum(s, sh);
    if (tid == 0) dis[i] = rsqrtf(s);
}

// ---------------- normalized A_rec output (zero diagonal) ----------------
__global__ __launch_bounds__(256) void k_arec_out(
    const float* __restrict__ Arec, const float* __restrict__ dis, float* __restrict__ out)
{
    const int i = blockIdx.x;
    const int tid = threadIdx.x;
    const float di = dis[i];
    for (int j = tid; j < K_CL; j += 256)
        out[(size_t)i * K_CL + j] = (i == j) ? 0.f : Arec[(size_t)i * K_CL + j] * di * dis[j];
}

extern "C" void kernel_launch(void* const* d_in, const int* in_sizes, int n_in,
                              void* d_out, int out_size, void* d_ws, size_t ws_size,
                              hipStream_t stream)
{
    (void)in_sizes; (void)n_in; (void)out_size; (void)ws_size;
    const float* X  = (const float*)d_in[0];
    const float* A  = (const float*)d_in[1];
    // d_in[2] = I (all zeros; single graph) — unused
    const float* W1 = (const float*)d_in[3];
    const float* b1 = (const float*)d_in[4];
    const float* W2 = (const float*)d_in[5];
    const float* b2 = (const float*)d_in[6];
    float* out = (float*)d_out;

    // workspace carve-out (~186 MiB)
    char* ws = (char*)d_ws;
    size_t off = 0;
    auto carve = [&](size_t bytes) -> char* {
        char* p = ws + off;
        off = (off + bytes + 255) & ~(size_t)255;
        return p;
    };
    __bf16* Ab  = (__bf16*)carve((size_t)N_NODES * N_NODES * 2);  // 128 MB (L2-resident)
    __bf16* Sb  = (__bf16*)carve((size_t)N_NODES * K_CL * 2);     // S    [N,k]
    __bf16* STb = (__bf16*)carve((size_t)K_CL * N_NODES * 2);     // S^T  [k,N]
    __bf16* ASt = (__bf16*)carve((size_t)K_CL * N_NODES * 2);     // (A@S)^T [k,N]
    __bf16* Xb  = (__bf16*)carve((size_t)N_NODES * F_DIM * 2);    // X bf16  [N,F]
    __bf16* XbT = (__bf16*)carve((size_t)F_DIM * N_NODES * 2);    // X^T bf16 [F,N]
    float*  Arec = (float*)carve((size_t)K_CL * K_CL * 4);        // S^T A S [k,k]
    float*  SSm  = (float*)carve((size_t)K_CL * K_CL * 4);        // S^T S   [k,k]
    float*  dsum = (float*)carve((size_t)N_NODES * 4);            // degrees
    float*  rSS  = (float*)carve((size_t)N_NODES * 4);            // row sum S*S
    float*  dis  = (float*)carve((size_t)K_CL * 4);               // deg^-0.5

    float* outX   = out;                          // [1024,64]
    float* outA   = out + (size_t)K_CL * F_DIM;   // [1024,1024]
    float* outI   = outA + (size_t)K_CL * K_CL;   // [1024]
    float* outCut = outI + K_CL;                  // [1]
    float* outOrt = outCut + 1;                   // [1]

    k_convert_rowsum<<<N_NODES, 256, 0, stream>>>(A, Ab, dsum);
    k_assign<<<N_NODES, 256, 0, stream>>>(X, W1, b1, W2, b2, Sb, Xb, rSS);
    k_transpose_bf16<<<dim3(N_NODES / 64, K_CL / 64), 256, 0, stream>>>(Sb, STb, N_NODES, K_CL);
    k_transpose_bf16<<<dim3(N_NODES / 64, F_DIM / 64), 256, 0, stream>>>(Xb, XbT, N_NODES, F_DIM);

    // (A@S)^T = (A @ S) stored bf16-transposed; B given as S^T [k,N]
    k_gemm_bf16<false, 1><<<dim3(N_NODES / 128, K_CL / 128), 256, 0, stream>>>(
        Ab, STb, nullptr, ASt, N_NODES, K_CL, N_NODES);
    // A_rec = S^T @ (A@S); B given as (A@S)^T
    k_gemm_bf16<false, 0><<<dim3(K_CL / 128, K_CL / 128), 256, 0, stream>>>(
        STb, ASt, Arec, nullptr, K_CL, K_CL, N_NODES);
    // SS = S^T @ S; B given as S^T
    k_gemm_bf16<false, 0><<<dim3(K_CL / 128, K_CL / 128), 256, 0, stream>>>(
        STb, STb, SSm, nullptr, K_CL, K_CL, N_NODES);
    // X_rec = S^T @ X; B given as X^T [64,N], N=64 -> row-guarded tile loads
    k_gemm_bf16<true, 0><<<dim3(K_CL / 128, 1), 256, 0, stream>>>(
        STb, XbT, outX, nullptr, K_CL, F_DIM, N_NODES);

    k_scalars<<<1, 1024, 0, stream>>>(Arec, SSm, dsum, rSS, outCut, outOrt, outI);
    k_dis<<<K_CL, 256, 0, stream>>>(Arec, dis);
    k_arec_out<<<K_CL, 256, 0, stream>>>(Arec, dis, outA);
}
